// LSTMTagger_18562848653669
// MI455X (gfx1250) — matrix operand order repaired and checked by reading, not compiled, not run on hardware
//
#include <hip/hip_runtime.h>
#include <stdint.h>

typedef __attribute__((ext_vector_type(16))) _Float16 v16h;
typedef __attribute__((ext_vector_type(8)))  float    v8f;

#define L_STEPS 262144
#define HID     128
#define GATES   512
#define CHUNK   256   // timesteps of x staged in LDS per refill (CHUNK*2 floats = 2KB)
#define NCHUNK  (L_STEPS / CHUNK)

__device__ __forceinline__ float fast_tanh(float x) {
#if __has_builtin(__builtin_amdgcn_tanhf)
    return __builtin_amdgcn_tanhf(x);            // v_tanh_f32 (CDNA5 TRANS op)
#elif __has_builtin(__builtin_amdgcn_tanh_f32)
    return __builtin_amdgcn_tanh_f32(x);
#else
    // Direct TRANS-unit tanh; v_nop inside covers the TRANS RAW hazard the
    // compiler cannot see through the asm.
    float r;
    asm volatile("v_tanh_f32 %0, %1\n\tv_nop" : "=v"(r) : "v"(x));
    return r;
#endif
}

// sigmoid(x) = 0.5*tanh(x/2) + 0.5 : one TRANS op + one FMA, no division.
__device__ __forceinline__ float sigmoidf(float x) {
    return __builtin_fmaf(0.5f, fast_tanh(0.5f * x), 0.5f);
}

// One persistent workgroup (16 wave32 waves) runs the whole recurrence.
// W_hh lives in WMMA A-layout f16 VGPRs for the entire loop; h is broadcast
// into the B operand each step, so all 16 WMMA output columns hold the matvec.
// Wave 15 double-buffers the input stream into LDS with async-to-LDS copies.
__launch_bounds__(512, 1)
__global__ void lstm_recurrence_kernel(
    const float* __restrict__ sentence, // (L,1,2)
    const float* __restrict__ W_ih,     // (512,2)
    const float* __restrict__ W_hh,     // (512,128)
    const float* __restrict__ b_ih,     // (512)
    const float* __restrict__ b_hh,     // (512)
    const float* __restrict__ W_out,    // (1,128)
    const float* __restrict__ b_out,    // (1)
    float* __restrict__ out)            // [sig_out(1) | h_n(128) | c_n(128)]
{
    __shared__ _Float16 h16[HID];          // h in f16 for the B operand
    __shared__ float    hW[GATES];         // per-step W_hh @ h result
    __shared__ float    xs[2][CHUNK * 2];  // double-buffered input chunks

    const int tid  = threadIdx.x;
    const int lane = tid & 31;
    const int wave = tid >> 5;          // 0..15
    const int mrow = lane & 15;         // A-layout: row within tile = lane%16
    const int hi   = lane >> 4;         // half-wave selects K group

    // ---- one-time: preload W_hh into WMMA A-layout f16 registers ----
    // wave w owns M-tiles {2w, 2w+1} (gate rows 32w .. 32w+31), K = 0..127.
    v16h A[2][4];
    #pragma unroll
    for (int tile = 0; tile < 2; ++tile) {
        const int row = wave * 32 + tile * 16 + mrow;
        #pragma unroll
        for (int kc = 0; kc < 4; ++kc) {
            #pragma unroll
            for (int v = 0; v < 8; ++v) {
                const int kb = kc * 32 + (v < 4 ? 2 * v : 16 + 2 * (v - 4)) + 8 * hi;
                A[tile][kc][2 * v]     = (_Float16)W_hh[row * HID + kb];
                A[tile][kc][2 * v + 1] = (_Float16)W_hh[row * HID + kb + 1];
            }
        }
    }
    // Pin the converted tiles as opaque register tuples so the compiler keeps
    // the f16 values live instead of re-deriving them from f32 each iteration.
    #pragma unroll
    for (int tile = 0; tile < 2; ++tile)
        #pragma unroll
        for (int kc = 0; kc < 4; ++kc)
            asm volatile("" : "+v"(A[tile][kc]));

    // ---- gate-thread state: thread j (<128) owns hidden unit j ----
    float wi0[4], wi1[4], bb[4];
    float hreg = 0.0f, creg = 0.0f;
    if (tid < HID) {
        #pragma unroll
        for (int g = 0; g < 4; ++g) {
            const int r = g * HID + tid;
            wi0[g] = W_ih[r * 2 + 0];
            wi1[g] = W_ih[r * 2 + 1];
            bb[g]  = b_ih[r] + b_hh[r];
        }
        h16[tid] = (_Float16)0.0f;      // h0 = 0
    }

    // ---- prologue: chunk 0 synchronously, chunk 1 via async-to-LDS ----
    xs[0][tid] = sentence[tid];
    if (wave == 15) {
        const uint32_t ldsdst = (uint32_t)(uintptr_t)(&xs[1][0]) + (uint32_t)lane * 16u;
        const uint32_t voff   = (uint32_t)lane * 16u;
        const uint64_t sbase  = (uint64_t)(uintptr_t)(sentence + CHUNK * 2);
        asm volatile(
            "global_load_async_to_lds_b128 %0, %1, %2 offset:0\n\t"
            "global_load_async_to_lds_b128 %0, %1, %2 offset:512\n\t"
            "global_load_async_to_lds_b128 %0, %1, %2 offset:1024\n\t"
            "global_load_async_to_lds_b128 %0, %1, %2 offset:1536"
            :: "v"(ldsdst), "v"(voff), "s"(sbase) : "memory");
    }
    __syncthreads();

    for (int c = 0; c < NCHUNK; ++c) {
        // ---- chunk boundary (outside the hot loop): retire the in-flight
        //      async copy for chunk c, start the copy for chunk c+1 ----
        if (c > 0 && wave == 15) {
            asm volatile("s_wait_asynccnt 0x0" ::: "memory");
            const int nc = c + 1;
            if (nc < NCHUNK) {
                const uint32_t ldsdst =
                    (uint32_t)(uintptr_t)(&xs[nc & 1][0]) + (uint32_t)lane * 16u;
                const uint32_t voff  = (uint32_t)lane * 16u;
                const uint64_t sbase = (uint64_t)(uintptr_t)(sentence + nc * CHUNK * 2);
                asm volatile(
                    "global_load_async_to_lds_b128 %0, %1, %2 offset:0\n\t"
                    "global_load_async_to_lds_b128 %0, %1, %2 offset:512\n\t"
                    "global_load_async_to_lds_b128 %0, %1, %2 offset:1024\n\t"
                    "global_load_async_to_lds_b128 %0, %1, %2 offset:1536"
                    :: "v"(ldsdst), "v"(voff), "s"(sbase) : "memory");
            }
            // visibility for other waves: first mid-step barrier of this chunk
        }
        const float* xbuf = &xs[c & 1][0];

        for (int ts = 0; ts < CHUNK; ++ts) {
            // ---- build broadcast B operand: every lane of a half holds h
            //      pairs, so all 16 output columns compute the same matvec ----
            v16h B[4];
            #pragma unroll
            for (int kc = 0; kc < 4; ++kc) {
                #pragma unroll
                for (int v = 0; v < 8; ++v) {
                    const int kb = kc * 32 + (v < 4 ? 2 * v : 16 + 2 * (v - 4)) + 8 * hi;
                    B[kc][2 * v]     = h16[kb];
                    B[kc][2 * v + 1] = h16[kb + 1];
                }
            }

            // ---- 2 M-tiles x 4 K-chunks of v_wmma_f32_16x16x32_f16 ----
            #pragma unroll
            for (int tile = 0; tile < 2; ++tile) {
                v8f acc = {};
                #pragma unroll
                for (int kc = 0; kc < 4; ++kc)
                    acc = __builtin_amdgcn_wmma_f32_16x16x32_f16(
                              false, A[tile][kc], false, B[kc],
                              (short)0, acc, false, false);
                // D layout: lane with N==0 holds rows (8*hi + r), r = VGPR 0..7
                if ((lane & 15) == 0) {
                    float* dst = &hW[wave * 32 + tile * 16 + hi * 8];
                    #pragma unroll
                    for (int r = 0; r < 8; ++r) dst[r] = acc[r];
                }
            }
            __syncthreads();

            // ---- gate nonlinearities + state update (threads 0..127) ----
            if (tid < HID) {
                const float x0 = xbuf[ts * 2 + 0];
                const float x1 = xbuf[ts * 2 + 1];
                const float gi = hW[tid      ] + x0 * wi0[0] + x1 * wi1[0] + bb[0];
                const float gf = hW[tid + 128] + x0 * wi0[1] + x1 * wi1[1] + bb[1];
                const float gg = hW[tid + 256] + x0 * wi0[2] + x1 * wi1[2] + bb[2];
                const float go = hW[tid + 384] + x0 * wi0[3] + x1 * wi1[3] + bb[3];
                const float i  = sigmoidf(gi);
                const float f  = sigmoidf(gf);
                const float o  = sigmoidf(go);
                const float gc = fast_tanh(gg);
                creg = f * creg + i * gc;
                hreg = o * fast_tanh(creg);
                h16[tid] = (_Float16)hreg;
            }
            __syncthreads();
        }
    }

    // ---- epilogue: only last-step outputs are needed ----
    if (tid < HID) {
        hW[tid] = W_out[tid] * hreg;    // reuse hW for the output reduction
        out[1 + tid]   = hreg;          // h_n
        out[129 + tid] = creg;          // c_n
    }
    __syncthreads();
    if (tid == 0) {
        float s = 0.0f;
        for (int j = 0; j < HID; ++j) s += hW[j];
        out[0] = sigmoidf(s + b_out[0]);
    }
}

extern "C" void kernel_launch(void* const* d_in, const int* in_sizes, int n_in,
                              void* d_out, int out_size, void* d_ws, size_t ws_size,
                              hipStream_t stream) {
    (void)in_sizes; (void)n_in; (void)out_size; (void)d_ws; (void)ws_size;
    const float* sentence = (const float*)d_in[0];
    const float* W_ih     = (const float*)d_in[1];
    const float* W_hh     = (const float*)d_in[2];
    const float* b_ih     = (const float*)d_in[3];
    const float* b_hh     = (const float*)d_in[4];
    const float* W_out    = (const float*)d_in[5];
    const float* b_out    = (const float*)d_in[6];

    // Strictly sequential recurrence -> one persistent workgroup on one WGP:
    // 512 threads = 16 wave32 waves, W_hh resident in WMMA A-layout VGPRs.
    lstm_recurrence_kernel<<<1, 512, 0, stream>>>(
        sentence, W_ih, W_hh, b_ih, b_hh, W_out, b_out, (float*)d_out);
}
